// MultiheadKANAttention_30666066494088
// MI455X (gfx1250) — matrix operand order repaired and checked by reading, not compiled, not run on hardware
//
#include <hip/hip_runtime.h>
#include <hip/hip_bf16.h>

// ---------------------------------------------------------------------------
// MultiheadKANAttention for MI455X (gfx1250, wave32, WMMA).
//
// Pipeline:
//   1) cheb_basis   : A[4096 x 3072] f16 = Chebyshev basis of tanh(x)
//   2) pack_wqkv    : Wt[2304 x 3072] f16 = cheby_coeffs repacked N-major
//   3) wmma_gemm    : qkv[4096 x 2304] f16 = A @ W   (v_wmma_f32_16x16x32_f16,
//                     async global->LDS double-buffered tiles)
//   4) rope_split   : q,k [b,h,s,64] f16 (RoPE applied; q pre-scaled by 1/8),
//                     vT [b,h,64,s] f16
//   5) flash_attn   : ctx[4096 x 768] f16, online-softmax attention via WMMA,
//                     async double-buffered K/V tiles, ds_swizzle butterflies
//   6) wmma_gemm    : out[4096 x 768] f32 = ctx @ out_w^T + out_b
// ---------------------------------------------------------------------------

typedef __attribute__((ext_vector_type(16))) _Float16 v16h;
typedef __attribute__((ext_vector_type(8)))  _Float16 v8h;
typedef __attribute__((ext_vector_type(8)))  float    v8f;

#define NUM_HEADS 12
#define HEAD_DIM  64
#define SEQ       2048
#define HID       768
#define NTOK      4096                 // B*S
#define KQKV      3072                 // HID * (DEGREE+1)
#define OQKV      2304                 // 3 * HID

union FragU { v16h v; v8h h2[2]; };

// ---------------------------------------------------------------------------
// CDNA5 async global->LDS copy (GLOBAL_LOAD_ASYNC_TO_LDS_B128, ASYNCcnt).
// LDS destination address = low 32 bits of the generic pointer.
// ---------------------------------------------------------------------------
__device__ __forceinline__ void async_copy_b128(const void* gsrc, void* ldst)
{
    unsigned           lds = (unsigned)(unsigned long long)ldst;
    unsigned long long g   = (unsigned long long)gsrc;
    asm volatile("global_load_async_to_lds_b128 %0, %1, off"
                 :: "v"(lds), "v"(g) : "memory");
}
__device__ __forceinline__ void wait_async()
{
    asm volatile("s_wait_asynccnt 0x0" ::: "memory");
}

// ---------------------------------------------------------------------------
// Single-instruction XOR butterfly via DS_SWIZZLE_B32 (group-of-32 mode):
// offset = {xor_mask[4:0] << 10, or_mask=0 << 5, and_mask=0x1F}.
// Masks < 16 keep exchanges within each 16-lane half (matches WMMA C-layout).
// ---------------------------------------------------------------------------
template<int XMASK>
__device__ __forceinline__ float swz_xor(float x)
{
    int i = __builtin_bit_cast(int, x);
    i = __builtin_amdgcn_ds_swizzle(i, (XMASK << 10) | 0x1F);
    return __builtin_bit_cast(float, i);
}
__device__ __forceinline__ float hmax16(float v)
{
    v = fmaxf(v, swz_xor<1>(v));
    v = fmaxf(v, swz_xor<2>(v));
    v = fmaxf(v, swz_xor<4>(v));
    v = fmaxf(v, swz_xor<8>(v));
    return v;
}
__device__ __forceinline__ float hsum16(float v)
{
    v += swz_xor<1>(v);
    v += swz_xor<2>(v);
    v += swz_xor<4>(v);
    v += swz_xor<8>(v);
    return v;
}

// ---------------------------------------------------------------------------
// 1) Chebyshev basis: A[row][i*4+j] = T_j(tanh(x[row,i])), T1 = 2x variant.
// ---------------------------------------------------------------------------
__global__ __launch_bounds__(256) void cheb_basis_kernel(
    const float* __restrict__ x, _Float16* __restrict__ A)
{
    int idx = blockIdx.x * blockDim.x + threadIdx.x;   // 4096*768 threads
    float t  = tanhf(x[idx]);
    float t0 = 1.0f;
    float t1 = 2.0f * t;
    float t2 = 2.0f * t * t1 - t0;
    float t3 = 2.0f * t * t2 - t1;
    int row = idx / HID, i = idx % HID;
    _Float16* p = A + (size_t)row * KQKV + i * 4;
    p[0] = (_Float16)t0; p[1] = (_Float16)t1;
    p[2] = (_Float16)t2; p[3] = (_Float16)t3;
}

// ---------------------------------------------------------------------------
// 2) Repack cheby_coeffs (768 x 2304 x 4 f32) -> Wt[o][i*4+j] f16 (N x K).
// ---------------------------------------------------------------------------
__global__ __launch_bounds__(256) void pack_wqkv_kernel(
    const float* __restrict__ C, _Float16* __restrict__ Wt)
{
    int idx = blockIdx.x * blockDim.x + threadIdx.x;   // 768*2304 threads
    int i = idx / OQKV, o = idx % OQKV;
    const float* src = C + ((size_t)i * OQKV + o) * 4;
    _Float16* dst = Wt + (size_t)o * KQKV + i * 4;
    dst[0] = (_Float16)src[0]; dst[1] = (_Float16)src[1];
    dst[2] = (_Float16)src[2]; dst[3] = (_Float16)src[3];
}

// ---------------------------------------------------------------------------
// f32 -> f16 elementwise (out_w is already in Bt = [n][k] layout).
// ---------------------------------------------------------------------------
__global__ __launch_bounds__(256) void f32_to_f16_kernel(
    const float* __restrict__ src, _Float16* __restrict__ dst)
{
    int idx = blockIdx.x * blockDim.x + threadIdx.x;
    dst[idx] = (_Float16)src[idx];
}

// ---------------------------------------------------------------------------
// WMMA GEMM: C[M x N] = A[M x K] * B, with B given transposed: Bt[N x K].
// Block tile 128x128, 8 waves, each wave computes 64x32 (4x2 of 16x16).
// A-fragment per-lane: lane<16 -> row=lane,  K = 0..15 sequential
//                      lane>=16 -> row=l-16, K = 16..31 sequential
// B-fragment per-lane: lane<16 -> col=lane,  K = 0..15 sequential
//                      lane>=16 -> col=l-16, K = 16..31 sequential
// Tiles are double-buffered; next tile streams in via async global->LDS
// while the current tile is consumed by WMMA.
// ---------------------------------------------------------------------------
template<bool HALF_OUT>
__global__ __launch_bounds__(256) void wmma_gemm_kernel(
    const _Float16* __restrict__ A, const _Float16* __restrict__ Bt,
    float* __restrict__ Cf, _Float16* __restrict__ Ch,
    const float* __restrict__ bias, int M, int N, int K)
{
    __shared__ alignas(16) _Float16 As[2][128][32];
    __shared__ alignas(16) _Float16 Bs[2][128][32];

    const int tid  = threadIdx.x;
    const int lane = tid & 31;
    const int wave = tid >> 5;
    const int l16  = lane & 15;
    const bool hi  = lane >= 16;
    const int bm = blockIdx.x * 128;
    const int bn = blockIdx.y * 128;
    const int wm = (wave & 1) * 64;     // 2 row-groups of 64
    const int wn = (wave >> 1) * 32;    // 4 col-groups of 32

    v8f acc[4][2] = {};
    const int nk = K / 32;

    // Prologue: async-stage tile 0 into buffer 0.
    for (int c = tid; c < 512; c += 256) {
        int r = c >> 2, kk = (c & 3) * 8;
        async_copy_b128(A  + (size_t)(bm + r) * K + kk, &As[0][r][kk]);
        async_copy_b128(Bt + (size_t)(bn + r) * K + kk, &Bs[0][r][kk]);
    }
    wait_async();
    __syncthreads();

    for (int kt = 0; kt < nk; ++kt) {
        const int buf = kt & 1;
        // Stream next tile into the other buffer (overlaps with WMMA below).
        if (kt + 1 < nk) {
            const int k0n = (kt + 1) * 32;
            for (int c = tid; c < 512; c += 256) {
                int r = c >> 2, kk = (c & 3) * 8;
                async_copy_b128(A  + (size_t)(bm + r) * K + k0n + kk, &As[buf ^ 1][r][kk]);
                async_copy_b128(Bt + (size_t)(bn + r) * K + k0n + kk, &Bs[buf ^ 1][r][kk]);
            }
        }

        v16h bf[2];
        {
            const int kb = hi ? 16 : 0;
#pragma unroll
            for (int j = 0; j < 2; ++j) {
                FragU u; const int n = wn + j * 16 + l16;
                u.h2[0] = *(const v8h*)&Bs[buf][n][kb];
                u.h2[1] = *(const v8h*)&Bs[buf][n][kb + 8];
                bf[j] = u.v;
            }
        }
#pragma unroll
        for (int i = 0; i < 4; ++i) {
            FragU u; const int r = wm + i * 16 + l16; const int kb = hi ? 16 : 0;
            u.h2[0] = *(const v8h*)&As[buf][r][kb];
            u.h2[1] = *(const v8h*)&As[buf][r][kb + 8];
#pragma unroll
            for (int j = 0; j < 2; ++j)
                acc[i][j] = __builtin_amdgcn_wmma_f32_16x16x32_f16(
                    false, u.v, false, bf[j], (short)0, acc[i][j], false, false);
        }
        // Next tile landed + all waves done reading this buffer.
        wait_async();
        __syncthreads();
    }

#pragma unroll
    for (int i = 0; i < 4; ++i)
#pragma unroll
        for (int j = 0; j < 2; ++j) {
            const int n = bn + wn + j * 16 + l16;
            const float bv = bias ? bias[n] : 0.0f;
#pragma unroll
            for (int p = 0; p < 8; ++p) {
                const int m = bm + wm + i * 16 + p + (hi ? 8 : 0);
                const float v = acc[i][j][p] + bv;
                if constexpr (HALF_OUT) Ch[(size_t)m * N + n] = (_Float16)v;
                else                    Cf[(size_t)m * N + n] = v;
            }
        }
}

// ---------------------------------------------------------------------------
// 4) RoPE + qkv split. One thread per (b,h,s,c) with c in [0,32).
//    q,k  -> [bh][s][64]  (A / Bt layouts for Q·K^T)
//    v    -> [bh][64][s]  (Bt layout for P·V)
// Q is pre-scaled by 1/sqrt(HEAD_DIM) = 0.125 (exact power of two), so the
// attention kernel needs no per-score multiply.
// ---------------------------------------------------------------------------
__global__ __launch_bounds__(256) void rope_split_kernel(
    const _Float16* __restrict__ qkv, const float* __restrict__ pos,
    _Float16* __restrict__ q, _Float16* __restrict__ k, _Float16* __restrict__ vt)
{
    int idx = blockIdx.x * blockDim.x + threadIdx.x;   // 24*2048*32
    int c  = idx & 31;
    int s  = (idx >> 5) & 2047;
    int bh = idx >> 16;
    if (bh >= 24) return;
    int b = bh / NUM_HEADS, h = bh % NUM_HEADS;

    size_t tok = (size_t)b * SEQ + s;
    const _Float16* base = qkv + tok * OQKV + h * 192;
    float cv  = pos[s * 64 + c];
    float sv  = pos[s * 64 + 32 + c];
    int   cp  = (c < 16) ? (c + 16) : (c - 16);   // rotate_half partner
    float sgn = (c < 16) ? -1.0f : 1.0f;
    size_t hb = ((size_t)bh * SEQ + s) * 64;

    { // q (pre-scaled by 1/8)
        float xe  = (float)base[2 * c],      xo  = (float)base[2 * c + 1];
        float xep = (float)base[2 * cp],     xop = (float)base[2 * cp + 1];
        q[hb + c]      = (_Float16)(0.125f * (xe * cv + sgn * xop * sv));
        q[hb + 32 + c] = (_Float16)(0.125f * (xo * cv + sgn * xep * sv));
    }
    { // k
        const _Float16* kb = base + 64;
        float xe  = (float)kb[2 * c],        xo  = (float)kb[2 * c + 1];
        float xep = (float)kb[2 * cp],       xop = (float)kb[2 * cp + 1];
        k[hb + c]      = (_Float16)(xe * cv + sgn * xop * sv);
        k[hb + 32 + c] = (_Float16)(xo * cv + sgn * xep * sv);
    }
    { // v, transposed [d][s]
        const _Float16* vb = base + 128;
        size_t vbase = (size_t)bh * 64 * SEQ;
        vt[vbase + (size_t)c        * SEQ + s] = vb[c];
        vt[vbase + (size_t)(c + 32) * SEQ + s] = vb[c + 32];
    }
}

// ---------------------------------------------------------------------------
// 5) Flash attention. Grid (S/128, B*NH), 256 threads = 8 waves.
//    Each wave owns 16 query rows => per-wave online softmax, no cross-wave
//    reduction. K/V tiles of 64 double-buffered in LDS via async copies;
//    P round-trips through LDS to convert WMMA C-layout -> A-fragment layout.
//    Softmax row reductions use single-instruction ds_swizzle XOR butterflies.
// ---------------------------------------------------------------------------
__global__ __launch_bounds__(256) void flash_attn_kernel(
    const _Float16* __restrict__ Q, const _Float16* __restrict__ Kg,
    const _Float16* __restrict__ Vt, _Float16* __restrict__ ctx, int S)
{
    __shared__ alignas(16) _Float16 Ks[2][64][64];  // [s_k][d]
    __shared__ alignas(16) _Float16 Vs[2][64][64];  // [d][s_k]
    __shared__ alignas(16) _Float16 Ps[8][16][64];  // per-wave P tile

    const int tid  = threadIdx.x;
    const int lane = tid & 31;
    const int wave = tid >> 5;
    const int l16  = lane & 15;
    const bool hi  = lane >= 16;
    const int bh   = blockIdx.y;                 // 0..23
    const int b    = bh / NUM_HEADS, h = bh % NUM_HEADS;
    const int q0   = blockIdx.x * 128 + wave * 16;
    const size_t head = (size_t)bh * S * 64;

    // Q A-fragments (K dim = d, two 32-wide chunks), loaded once.
    v16h aq[2];
    {
        const _Float16* qr = Q + head + (size_t)(q0 + l16) * 64;
        const int kb = hi ? 16 : 0;
        FragU u0, u1;
        u0.h2[0] = *(const v8h*)(qr + kb);      u0.h2[1] = *(const v8h*)(qr + kb + 8);
        u1.h2[0] = *(const v8h*)(qr + 32 + kb); u1.h2[1] = *(const v8h*)(qr + 32 + kb + 8);
        aq[0] = u0.v; aq[1] = u1.v;
    }

    float Mx[8], L[8];
#pragma unroll
    for (int p = 0; p < 8; ++p) { Mx[p] = -1e30f; L[p] = 0.0f; }
    v8f oacc[4] = {};

    const _Float16* Kbase = Kg + head;
    const _Float16* Vbase = Vt + (size_t)bh * 64 * S;
    const int nt = S / 64;

    // Prologue: async-stage tile 0.
    for (int c = tid; c < 512; c += 256) {
        int r = c >> 3, kk = (c & 7) * 8;
        async_copy_b128(Kbase + (size_t)r * 64 + kk, &Ks[0][r][kk]);
        async_copy_b128(Vbase + (size_t)r * S  + kk, &Vs[0][r][kk]);
    }
    wait_async();
    __syncthreads();

    for (int it = 0; it < nt; ++it) {
        const int buf = it & 1;
        if (it + 1 < nt) {
            const int s0n = (it + 1) * 64;
            for (int c = tid; c < 512; c += 256) {
                int r = c >> 3, kk = (c & 7) * 8;
                async_copy_b128(Kbase + (size_t)(s0n + r) * 64 + kk, &Ks[buf ^ 1][r][kk]);
                async_copy_b128(Vbase + (size_t)r * S + s0n + kk,    &Vs[buf ^ 1][r][kk]);
            }
        }

        // S = Q K^T (Q pre-scaled): 16 x 64 in 4 col-subtiles.
        v8f sf[4];
#pragma unroll
        for (int j = 0; j < 4; ++j) {
            const int n  = j * 16 + l16;
            const int kb = hi ? 16 : 0;
            FragU b0, b1;
            b0.h2[0] = *(const v8h*)&Ks[buf][n][kb];      b0.h2[1] = *(const v8h*)&Ks[buf][n][kb + 8];
            b1.h2[0] = *(const v8h*)&Ks[buf][n][32 + kb]; b1.h2[1] = *(const v8h*)&Ks[buf][n][32 + kb + 8];
            v8f z = {};
            z = __builtin_amdgcn_wmma_f32_16x16x32_f16(false, aq[0], false, b0.v, (short)0, z, false, false);
            z = __builtin_amdgcn_wmma_f32_16x16x32_f16(false, aq[1], false, b1.v, (short)0, z, false, false);
            sf[j] = z;
        }

        // Online softmax (rows live in VGPR index p, 16-lane halves independent).
        float mnew[8], corr[8];
#pragma unroll
        for (int p = 0; p < 8; ++p) {
            float lm = hmax16(fmaxf(fmaxf(sf[0][p], sf[1][p]), fmaxf(sf[2][p], sf[3][p])));
            mnew[p] = fmaxf(Mx[p], lm);
            corr[p] = __expf(Mx[p] - mnew[p]);
            Mx[p]   = mnew[p];
        }
#pragma unroll
        for (int j = 0; j < 4; ++j)
#pragma unroll
            for (int p = 0; p < 8; ++p)
                sf[j][p] = __expf(sf[j][p] - mnew[p]);
#pragma unroll
        for (int p = 0; p < 8; ++p) {
            float r = hsum16(sf[0][p] + sf[1][p] + sf[2][p] + sf[3][p]);
            L[p] = L[p] * corr[p] + r;
        }
#pragma unroll
        for (int j = 0; j < 4; ++j)
#pragma unroll
            for (int p = 0; p < 8; ++p)
                oacc[j][p] *= corr[p];

        // C-layout -> LDS -> A-fragment layout for P (per-wave region,
        // wave-internal DS ordering handled by compiler-inserted s_wait_dscnt).
        const int mr = hi ? 8 : 0;
#pragma unroll
        for (int j = 0; j < 4; ++j)
#pragma unroll
            for (int p = 0; p < 8; ++p)
                Ps[wave][mr + p][j * 16 + l16] = (_Float16)sf[j][p];

        v16h ap[2];
        {
            const _Float16* pr = &Ps[wave][l16][0];
            const int kb = hi ? 16 : 0;
            FragU u0, u1;
            u0.h2[0] = *(const v8h*)(pr + kb);      u0.h2[1] = *(const v8h*)(pr + kb + 8);
            u1.h2[0] = *(const v8h*)(pr + 32 + kb); u1.h2[1] = *(const v8h*)(pr + 32 + kb + 8);
            ap[0] = u0.v; ap[1] = u1.v;
        }

        // O += P @ V  (K dim = s_k, two 32-wide chunks).
#pragma unroll
        for (int j = 0; j < 4; ++j) {
            const int n  = j * 16 + l16;
            const int kb = hi ? 16 : 0;
            FragU b0, b1;
            b0.h2[0] = *(const v8h*)&Vs[buf][n][kb];      b0.h2[1] = *(const v8h*)&Vs[buf][n][kb + 8];
            b1.h2[0] = *(const v8h*)&Vs[buf][n][32 + kb]; b1.h2[1] = *(const v8h*)&Vs[buf][n][32 + kb + 8];
            oacc[j] = __builtin_amdgcn_wmma_f32_16x16x32_f16(false, ap[0], false, b0.v, (short)0, oacc[j], false, false);
            oacc[j] = __builtin_amdgcn_wmma_f32_16x16x32_f16(false, ap[1], false, b1.v, (short)0, oacc[j], false, false);
        }
        // Next K/V tile landed + all waves done with this buffer.
        wait_async();
        __syncthreads();
    }

    // Normalize and store ctx[token][h*64 + d] as f16.
#pragma unroll
    for (int p = 0; p < 8; ++p) L[p] = 1.0f / L[p];
    const int mr2 = hi ? 8 : 0;
#pragma unroll
    for (int j = 0; j < 4; ++j) {
        const int col = h * 64 + j * 16 + l16;
#pragma unroll
        for (int p = 0; p < 8; ++p) {
            const int srow = q0 + mr2 + p;
            const size_t tok = (size_t)b * SEQ + srow;
            ctx[tok * HID + col] = (_Float16)(oacc[j][p] * L[p]);
        }
    }
}

// ---------------------------------------------------------------------------
// Host launcher.
// ---------------------------------------------------------------------------
extern "C" void kernel_launch(void* const* d_in, const int* in_sizes, int n_in,
                              void* d_out, int out_size, void* d_ws, size_t ws_size,
                              hipStream_t stream)
{
    const float* x     = (const float*)d_in[0];   // 2*2048*768
    const float* pos   = (const float*)d_in[1];   // 2048*64
    const float* cheb  = (const float*)d_in[2];   // 768*2304*4
    const float* out_w = (const float*)d_in[3];   // 768*768
    const float* out_b = (const float*)d_in[4];   // 768
    float* out = (float*)d_out;                   // 4096*768 f32

    char* ws = (char*)d_ws;
    // Region 0: A (25.2MB), later reused for q/k/vT (3 x 6.3MB).
    _Float16* A   = (_Float16*)(ws);
    _Float16* q   = (_Float16*)(ws);
    _Float16* k   = (_Float16*)(ws + 6291456);
    _Float16* vt  = (_Float16*)(ws + 12582912);
    // Region 1: Wt (14.2MB), later reused for ctx (6.3MB) + wout (1.2MB).
    _Float16* Wt   = (_Float16*)(ws + 25165824);
    _Float16* ctx  = (_Float16*)(ws + 25165824);
    _Float16* wout = (_Float16*)(ws + 25165824 + 6291456);
    // Region 2: qkv f16 (18.9MB).
    _Float16* qkv  = (_Float16*)(ws + 25165824 + 14155776);

    cheb_basis_kernel<<<(NTOK * HID) / 256, 256, 0, stream>>>(x, A);
    pack_wqkv_kernel<<<(HID * OQKV) / 256, 256, 0, stream>>>(cheb, Wt);

    // qkv = A @ W : M=4096, N=2304, K=3072 (f16 out, no bias)
    wmma_gemm_kernel<true><<<dim3(NTOK / 128, OQKV / 128), 256, 0, stream>>>(
        A, Wt, nullptr, qkv, nullptr, NTOK, OQKV, KQKV);

    // After GEMM1, region 0/1 are free for reuse.
    rope_split_kernel<<<(24 * SEQ * 32) / 256, 256, 0, stream>>>(qkv, pos, q, k, vt);
    f32_to_f16_kernel<<<(HID * HID) / 256, 256, 0, stream>>>(out_w, wout);

    flash_attn_kernel<<<dim3(SEQ / 128, 2 * NUM_HEADS), 256, 0, stream>>>(
        q, k, vt, ctx, SEQ);

    // out = ctx @ out_w^T + out_b : M=4096, N=768, K=768 (f32 out, bias)
    wmma_gemm_kernel<false><<<dim3(NTOK / 128, HID / 128), 256, 0, stream>>>(
        ctx, wout, out, nullptr, out_b, NTOK, HID, HID);
}
